// DualMHTransGraphModel_84241488544066
// MI455X (gfx1250) — compile-verified, matrix operand order checked by reading, not run on hardware
//
#include <hip/hip_runtime.h>
#include <hip/hip_bf16.h>
#include <math.h>

#define DEV __device__ __forceinline__

static constexpr int Bn  = 64;
static constexpr int Nn  = 128;
static constexpr int Dn  = 128;
static constexpr int Hn  = 4;
static constexpr int FFNn = 128;
static constexpr float NEGV   = -9000000000000000.0f;
static constexpr float LRELU  = 0.2f;

typedef __attribute__((ext_vector_type(16))) _Float16 v16h;
typedef __attribute__((ext_vector_type(8)))  _Float16 v8h;
typedef __attribute__((ext_vector_type(8)))  float    v8f;

// ---------------------------------------------------------------------------
// WMMA helpers (gfx1250, wave32): D(16x16 f32) = A(16x32 f16) x B(32x16 f16) + C
// A fragment: lane<16 -> row M=lane, halves0..7 = K k0+0..7, halves8..15 = K k0+16..23
//             lane>=16 -> row M=lane-16, K k0+8..15 and k0+24..31
// B fragment mirrored with column N = lane%16.  C/D: n=lane%16, m=v+8*(lane>=16).
// ---------------------------------------------------------------------------
DEV v8f wmma_f16(v16h a, v16h b, v8f c) {
  return __builtin_amdgcn_wmma_f32_16x16x32_f16(false, a, false, b, (short)0, c,
                                                false, false);
}

DEV v16h frag_row(const _Float16* row, int k0, int lane) {
  const int off = (lane & 16) ? 8 : 0;
  v8h lo = *(const v8h*)(row + k0 + off);
  v8h hi = *(const v8h*)(row + k0 + off + 16);
  return __builtin_shufflevector(lo, hi, 0,1,2,3,4,5,6,7,8,9,10,11,12,13,14,15);
}

// C = A(16 x K) * B^T-laid-out(16 x K) accumulate; ldA/ldB in halves
DEV v8f gemm_tile(const _Float16* Abase, int ldA, const _Float16* Bbase, int ldB,
                  int K, int lane) {
  v8f acc = {0.f,0.f,0.f,0.f,0.f,0.f,0.f,0.f};
  const _Float16* ar = Abase + (size_t)(lane & 15) * ldA;
  const _Float16* br = Bbase + (size_t)(lane & 15) * ldB;
  for (int k0 = 0; k0 < K; k0 += 32)
    acc = wmma_f16(frag_row(ar, k0, lane), frag_row(br, k0, lane), acc);
  return acc;
}

DEV void store_tile_f32(float* out, int ld, int lane, v8f acc) {
  const int n = lane & 15, mo = (lane & 16) ? 8 : 0;
  for (int v = 0; v < 8; v++) out[(size_t)(v + mo) * ld + n] = acc[v];
}
DEV void store_tile_f16(_Float16* out, int ld, int lane, v8f acc) {
  const int n = lane & 15, mo = (lane & 16) ? 8 : 0;
  for (int v = 0; v < 8; v++) out[(size_t)(v + mo) * ld + n] = (_Float16)acc[v];
}
// write tile transposed: element (m,n) -> out[n*ld + m]; contiguous 8-half store
DEV void store_tile_f16_trans(_Float16* out, int ld, int lane, v8f acc) {
  const int n = lane & 15, mo = (lane & 16) ? 8 : 0;
  v8h h;
  for (int v = 0; v < 8; v++) h[v] = (_Float16)acc[v];
  *(v8h*)(out + (size_t)n * ld + mo) = h;
}

DEV float wave_sum(float v) {
  for (int s = 16; s; s >>= 1) v += __shfl_xor(v, s, 32);
  return v;
}
DEV float wave_max(float v) {
  for (int s = 16; s; s >>= 1) v = fmaxf(v, __shfl_xor(v, s, 32));
  return v;
}

// ---------------------------------------------------------------------------
// K0a: hidden -> f16 row-major, f16 transposed, and 4 pre-scaled f16 copies
//      hs[r] = hidden * a_r  (hoists all GAT A-operand scaling out of the
//      WMMA loop; each copy is reused 8x per batch)
// ---------------------------------------------------------------------------
__global__ void __launch_bounds__(256)
k_prep_hidden(const float* hidden,
              const float* a0, const float* a1, const float* a2, const float* a3,
              _Float16* hh, _Float16* ht, _Float16* hs) {
  size_t i = (size_t)blockIdx.x * 256 + threadIdx.x;
  if (i >= (size_t)Bn * Nn * Dn) return;
  float v = hidden[i];
  hh[i] = (_Float16)v;
  size_t b = i / (Nn * Dn), r = i % (Nn * Dn);
  size_t n = r / Dn, d = r % Dn;
  ht[b * Dn * Nn + d * Nn + n] = (_Float16)v;
  const size_t S = (size_t)Bn * Nn * Dn;
  hs[i]         = (_Float16)(v * a0[d]);
  hs[S + i]     = (_Float16)(v * a1[d]);
  hs[2 * S + i] = (_Float16)(v * a2[d]);
  hs[3 * S + i] = (_Float16)(v * a3[d]);
}

// K0b: transpose all five weight matrices to f16 (all are 128x128)
__global__ void __launch_bounds__(256)
k_prep_w(const float* wq, const float* wk, const float* wv,
         const float* w1, const float* w2,
         _Float16* wqT, _Float16* wkT, _Float16* wvT,
         _Float16* w1T, _Float16* w2T) {
  int i = blockIdx.x * 256 + threadIdx.x;
  if (i >= 5 * 128 * 128) return;
  int r = i / (128 * 128), j = i % (128 * 128);
  int row = j / 128, col = j % 128;
  float v;
  switch (r) {
    case 0: v = wq[col * 128 + row]; wqT[j] = (_Float16)v; break;
    case 1: v = wk[col * 128 + row]; wkT[j] = (_Float16)v; break;
    case 2: v = wv[col * 128 + row]; wvT[j] = (_Float16)v; break;
    case 3: v = w1[col * 128 + row]; w1T[j] = (_Float16)v; break;
    default: v = w2[col * 128 + row]; w2T[j] = (_Float16)v; break;
  }
}

// ---------------------------------------------------------------------------
// K1: GAT logits.  e_r = (H ⊙ a_r)_rows(i-tile) @ H_j^T ; att = select(adj, lrelu)
// block = (i_tile, b), 8 waves = 8 j-tiles.
// LDS: hidden[b] (32KB, B-panel) + 4 pre-scaled A-row tiles (16KB).
// Inner loop is pure ds_load_b128 + v_wmma (no per-chunk VALU).
// ---------------------------------------------------------------------------
__global__ void __launch_bounds__(256)
k_gat(const _Float16* hh, const _Float16* hs, const int* adj, float* att_logits) {
  __shared__ _Float16 sh[Nn * Dn];       // 32 KB : hidden[b] rows (B operand)
  __shared__ _Float16 sA[4 * 16 * Dn];   // 16 KB : 4 scaled variants of i-tile rows
  const int b = blockIdx.y, it = blockIdx.x, tid = threadIdx.x;
  const int ib = it * 16;

  const _Float16* hb = hh + (size_t)b * Nn * Dn;
  for (int idx = tid; idx < Nn * Dn / 8; idx += 256)
    ((v8h*)sh)[idx] = ((const v8h*)hb)[idx];
  // stage 4 scaled A-row tiles: sA[r][m][d] = hs[r][b][ib+m][d]
  const size_t S = (size_t)Bn * Nn * Dn;
  for (int idx = tid; idx < 4 * 16 * Dn / 8; idx += 256) {
    int r = idx / (16 * Dn / 8), rem = idx % (16 * Dn / 8);
    ((v8h*)sA)[idx] =
        ((const v8h*)(hs + (size_t)r * S + ((size_t)b * Nn + ib) * Dn))[rem];
  }
  __syncthreads();

  const int lane = tid & 31, w = tid >> 5;
  const int jb = w * 16;
  const _Float16* arow = sA + (size_t)(lane & 15) * Dn;   // +r*16*Dn per variant
  const _Float16* brow = sh + (size_t)(jb + (lane & 15)) * Dn;

  const int n = lane & 15, mo = (lane & 16) ? 8 : 0;
  const int* adjb = adj + ((size_t)b * Nn + ib) * Nn;
  __builtin_prefetch(adjb + (size_t)mo * Nn + jb, 0, 0);  // global_prefetch_b8

  v8f acc[4];
  for (int r = 0; r < 4; r++) acc[r] = (v8f){0.f,0.f,0.f,0.f,0.f,0.f,0.f,0.f};

  for (int k0 = 0; k0 < Dn; k0 += 32) {
    v16h bf = frag_row(brow, k0, lane);
    for (int r = 0; r < 4; r++)
      acc[r] = wmma_f16(frag_row(arow + (size_t)r * 16 * Dn, k0, lane), bf, acc[r]);
  }

  float* outb = att_logits + ((size_t)b * Nn + ib) * Nn;
  for (int v = 0; v < 8; v++) {
    int m = v + mo;
    int av = adjb[(size_t)m * Nn + jb + n];
    float val = NEGV;
    for (int r = 0; r < 4; r++)
      if (av == r + 1) { float e = acc[r][v]; val = (e > 0.f) ? e : LRELU * e; }
    outb[(size_t)m * Nn + jb + n] = val;
  }
}

// K2: row softmax over j, output f16 probabilities. wave per row.
__global__ void __launch_bounds__(256)
k_softmax(const float* logits, _Float16* atth) {
  int row = blockIdx.x * 8 + (threadIdx.x >> 5);  // [0, B*N)
  int lane = threadIdx.x & 31;
  const float* x0 = logits + (size_t)row * Nn;
  float x[4];
  for (int u = 0; u < 4; u++) x[u] = x0[lane + 32 * u];
  float mx = wave_max(fmaxf(fmaxf(x[0], x[1]), fmaxf(x[2], x[3])));
  float e[4], s = 0.f;
  for (int u = 0; u < 4; u++) { e[u] = expf(x[u] - mx); s += e[u]; }
  s = wave_sum(s);
  float inv = 1.f / s;
  _Float16* o = atth + (size_t)row * Nn;
  for (int u = 0; u < 4; u++) o[lane + 32 * u] = (_Float16)(e[u] * inv);
}

// K3: local = att @ hidden  (B operand = hidden^T f16)
__global__ void __launch_bounds__(256)
k_local(const _Float16* atth, const _Float16* ht, float* localF, _Float16* localH) {
  int b = blockIdx.y, it = blockIdx.x;
  int lane = threadIdx.x & 31, w = threadIdx.x >> 5;
  int ib = it * 16, dt = w * 16;
  const _Float16* A = atth + ((size_t)b * Nn + ib) * Nn;
  const _Float16* Bp = ht + (size_t)b * Dn * Nn + (size_t)dt * Nn;
  v8f acc = gemm_tile(A, Nn, Bp, Nn, Nn, lane);
  store_tile_f32(localF + ((size_t)b * Nn + ib) * Dn + dt, Dn, lane, acc);
  store_tile_f16(localH + ((size_t)b * Nn + ib) * Dn + dt, Dn, lane, acc);
}

// K4: a_ent[b] = sigmoid(local[b,-1] . wa + ba) + 1 (with ==1 nudge)
__global__ void __launch_bounds__(32)
k_aent(const float* localF, const float* wa, const float* ba, float* aent) {
  int b = blockIdx.x, lane = threadIdx.x;
  const float* l = localF + ((size_t)b * Nn + (Nn - 1)) * Dn;
  float s = 0.f;
  for (int u = 0; u < 4; u++) s += l[lane + 32 * u] * wa[lane + 32 * u];
  s = wave_sum(s);
  if (lane == 0) {
    float v = 1.f / (1.f + expf(-(s + ba[0]))) + 1.f;
    if (v == 1.f) v = 1.00001f;
    aent[b] = v;
  }
}

// K5: q,k,v = local @ W + bias.  q,k row-major f16; v stored transposed f16.
__global__ void __launch_bounds__(256)
k_qkv(const _Float16* localH,
      const _Float16* wqT, const _Float16* wkT, const _Float16* wvT,
      const float* bq, const float* bk, const float* bv,
      _Float16* qh, _Float16* kh, _Float16* vt) {
  int b = blockIdx.y, it = blockIdx.x;
  int lane = threadIdx.x & 31, w = threadIdx.x >> 5;
  int ib = it * 16, dt = w * 16;
  const _Float16* A = localH + ((size_t)b * Nn + ib) * Dn;
  v8f q = gemm_tile(A, Dn, wqT + (size_t)dt * Dn, Dn, Dn, lane);
  v8f k = gemm_tile(A, Dn, wkT + (size_t)dt * Dn, Dn, Dn, lane);
  v8f v = gemm_tile(A, Dn, wvT + (size_t)dt * Dn, Dn, Dn, lane);
  int n = lane & 15;
  float bqv = bq[dt + n], bkv = bk[dt + n], bvv = bv[dt + n];
  for (int u = 0; u < 8; u++) { q[u] += bqv; k[u] += bkv; v[u] += bvv; }
  store_tile_f16(qh + ((size_t)b * Nn + ib) * Dn + dt, Dn, lane, q);
  store_tile_f16(kh + ((size_t)b * Nn + ib) * Dn + dt, Dn, lane, k);
  store_tile_f16_trans(vt + (size_t)b * Dn * Nn + (size_t)dt * Nn + ib, Nn, lane, v);
}

// K6: per-head scores = q k^T / sqrt(D), masked.  Single K=32 WMMA chunk.
__global__ void __launch_bounds__(256)
k_scores(const _Float16* qh, const _Float16* kh, const int* seq_mask, float* sc) {
  int b = blockIdx.z, h = blockIdx.y, it = blockIdx.x;
  int lane = threadIdx.x & 31, w = threadIdx.x >> 5;
  int ib = it * 16, jb = w * 16;
  const _Float16* A  = qh + ((size_t)b * Nn + ib) * Dn + h * 32;
  const _Float16* Bp = kh + ((size_t)b * Nn + jb) * Dn + h * 32;
  v8f acc = gemm_tile(A, Dn, Bp, Dn, 32, lane);
  const float scale = 0.088388347648318447f;  // 1/sqrt(128)
  int n = lane & 15, mo = (lane & 16) ? 8 : 0;
  int jc = jb + n;
  bool masked = (seq_mask[(size_t)b * Nn + jc] == 0);
  float* out = sc + (((size_t)(b * Hn + h) * Nn + ib)) * Nn;
  for (int v = 0; v < 8; v++) {
    float s = acc[v] * scale;
    out[(size_t)(v + mo) * Nn + jc] = masked ? -1000000000.0f : s;
  }
}

// K7: entmax bisection (30 iters), wave per row, uniform branches.
__global__ void __launch_bounds__(256)
k_entmax(const float* sc, const float* aent, _Float16* alph) {
  int row = blockIdx.x * 8 + (threadIdx.x >> 5);  // [0, B*H*N)
  int lane = threadIdx.x & 31;
  int b = row / (Hn * Nn);
  const float* x0 = sc + (size_t)row * Nn;
  float am1 = aent[b] - 1.f;
  float inv = 1.f / am1;
  float x[4];
  for (int u = 0; u < 4; u++) x[u] = x0[lane + 32 * u] * am1;
  float mx = wave_max(fmaxf(fmaxf(x[0], x[1]), fmaxf(x[2], x[3])));
  float tau_lo = mx - 1.f;
  float tau_hi = mx - exp2f(-7.f * am1);        // (1/128)^(alpha-1)

  auto psum = [&](float tau, float* pout) -> float {
    float s = 0.f;
    for (int u = 0; u < 4; u++) {
      float t = x[u] - tau;
      t = t > 0.f ? t : 0.f;
      float p = (t > 0.f) ? exp2f(inv * log2f(t)) : 0.f;
      if (pout) pout[u] = p;
      s += p;
    }
    return wave_sum(s);
  };

  float f_lo = psum(tau_lo, nullptr) - 1.f;
  for (int i = 0; i < 30; i++) {
    float tau_m = 0.5f * (tau_lo + tau_hi);
    float f_m = psum(tau_m, nullptr) - 1.f;
    if (f_m * f_lo >= 0.f) { tau_lo = tau_m; f_lo = f_m; }
    else                   { tau_hi = tau_m; }
  }
  float pm[4];
  float s = psum(0.5f * (tau_lo + tau_hi), pm);
  float invs = 1.f / s;
  _Float16* o = alph + (size_t)row * Nn;
  for (int u = 0; u < 4; u++) o[lane + 32 * u] = (_Float16)(pm[u] * invs);
}

// K8: y = alpha @ v per head (B operand = v^T), row-major f32 + f16 out.
__global__ void __launch_bounds__(256)
k_y(const _Float16* alph, const _Float16* vt, float* yF, _Float16* yH) {
  int b = blockIdx.y, it = blockIdx.x;
  int lane = threadIdx.x & 31, w = threadIdx.x >> 5;
  int ib = it * 16, dt = w * 16, h = w >> 1;
  const _Float16* A  = alph + ((size_t)(b * Hn + h) * Nn + ib) * Nn;
  const _Float16* Bp = vt + (size_t)b * Dn * Nn + (size_t)dt * Nn;
  v8f acc = gemm_tile(A, Nn, Bp, Nn, Nn, lane);
  store_tile_f32(yF + ((size_t)b * Nn + ib) * Dn + dt, Dn, lane, acc);
  store_tile_f16(yH + ((size_t)b * Nn + ib) * Dn + dt, Dn, lane, acc);
}

// K9: t1 = relu(y @ w1 + b1)
__global__ void __launch_bounds__(256)
k_ffn1(const _Float16* yH, const _Float16* w1T, const float* b1, _Float16* t1) {
  int b = blockIdx.y, it = blockIdx.x;
  int lane = threadIdx.x & 31, w = threadIdx.x >> 5;
  int ib = it * 16, ft = w * 16;
  const _Float16* A = yH + ((size_t)b * Nn + ib) * Dn;
  v8f acc = gemm_tile(A, Dn, w1T + (size_t)ft * Dn, Dn, Dn, lane);
  int n = lane & 15;
  float bb = b1[ft + n];
  for (int u = 0; u < 8; u++) { float t = acc[u] + bb; acc[u] = t > 0.f ? t : 0.f; }
  store_tile_f16(t1 + ((size_t)b * Nn + ib) * FFNn + ft, FFNn, lane, acc);
}

// K10: y2 = t1 @ w2 + b2
__global__ void __launch_bounds__(256)
k_ffn2(const _Float16* t1, const _Float16* w2T, const float* b2, float* y2) {
  int b = blockIdx.y, it = blockIdx.x;
  int lane = threadIdx.x & 31, w = threadIdx.x >> 5;
  int ib = it * 16, dt = w * 16;
  const _Float16* A = t1 + ((size_t)b * Nn + ib) * FFNn;
  v8f acc = gemm_tile(A, FFNn, w2T + (size_t)dt * FFNn, FFNn, FFNn, lane);
  int n = lane & 15;
  float bb = b2[dt + n];
  for (int u = 0; u < 8; u++) acc[u] += bb;
  store_tile_f32(y2 + ((size_t)b * Nn + ib) * Dn + dt, Dn, lane, acc);
}

// K11: x = local + y + y2; layernorm over D. wave per row.
__global__ void __launch_bounds__(256)
k_final(const float* localF, const float* yF, const float* y2,
        const float* g, const float* bb, float* out) {
  int row = blockIdx.x * 8 + (threadIdx.x >> 5);  // [0, B*N)
  int lane = threadIdx.x & 31;
  size_t base = (size_t)row * Dn;
  float x[4], s = 0.f, q = 0.f;
  for (int u = 0; u < 4; u++) {
    size_t idx = base + lane + 32 * u;
    x[u] = localF[idx] + yF[idx] + y2[idx];
    s += x[u];
    q += x[u] * x[u];
  }
  s = wave_sum(s); q = wave_sum(q);
  float mu = s * (1.f / Dn);
  float var = q * (1.f / Dn) - mu * mu;
  float r = rsqrtf(var + 1e-5f);
  for (int u = 0; u < 4; u++) {
    int d = lane + 32 * u;
    out[base + d] = (x[u] - mu) * r * g[d] + bb[d];
  }
}

// ---------------------------------------------------------------------------
extern "C" void kernel_launch(void* const* d_in, const int* in_sizes, int n_in,
                              void* d_out, int out_size, void* d_ws, size_t ws_size,
                              hipStream_t stream) {
  const float* hidden = (const float*)d_in[0];
  const int*   adj    = (const int*)d_in[1];
  const float* a0 = (const float*)d_in[2];
  const float* a1 = (const float*)d_in[3];
  const float* a2 = (const float*)d_in[4];
  const float* a3 = (const float*)d_in[5];
  const float* wa = (const float*)d_in[6];
  const float* ba = (const float*)d_in[7];
  const float* wq = (const float*)d_in[8];
  const float* bq = (const float*)d_in[9];
  const float* wk = (const float*)d_in[10];
  const float* bk = (const float*)d_in[11];
  const float* wv = (const float*)d_in[12];
  const float* bv = (const float*)d_in[13];
  const float* w1 = (const float*)d_in[14];
  const float* b1 = (const float*)d_in[15];
  const float* w2 = (const float*)d_in[16];
  const float* b2 = (const float*)d_in[17];
  const float* ln_g = (const float*)d_in[18];
  const float* ln_b = (const float*)d_in[19];
  const int*   seq_mask = (const int*)d_in[20];
  float* out = (float*)d_out;

  size_t off = 0;
  auto alloc = [&](size_t bytes) -> void* {
    void* p = (char*)d_ws + off;
    off += (bytes + 255) & ~(size_t)255;
    return p;
  };
  _Float16* hh   = (_Float16*)alloc((size_t)Bn * Nn * Dn * 2);
  _Float16* ht   = (_Float16*)alloc((size_t)Bn * Dn * Nn * 2);
  _Float16* hs   = (_Float16*)alloc((size_t)4 * Bn * Nn * Dn * 2);
  _Float16* wqT  = (_Float16*)alloc((size_t)Dn * Dn * 2);
  _Float16* wkT  = (_Float16*)alloc((size_t)Dn * Dn * 2);
  _Float16* wvT  = (_Float16*)alloc((size_t)Dn * Dn * 2);
  _Float16* w1T  = (_Float16*)alloc((size_t)FFNn * Dn * 2);
  _Float16* w2T  = (_Float16*)alloc((size_t)Dn * FFNn * 2);
  float*    attL = (float*)alloc((size_t)Bn * Nn * Nn * 4);
  _Float16* attH = (_Float16*)alloc((size_t)Bn * Nn * Nn * 2);
  float*    locF = (float*)alloc((size_t)Bn * Nn * Dn * 4);
  _Float16* locH = (_Float16*)alloc((size_t)Bn * Nn * Dn * 2);
  float*    aent = (float*)alloc((size_t)Bn * 4);
  _Float16* qh   = (_Float16*)alloc((size_t)Bn * Nn * Dn * 2);
  _Float16* kh   = (_Float16*)alloc((size_t)Bn * Nn * Dn * 2);
  _Float16* vt   = (_Float16*)alloc((size_t)Bn * Dn * Nn * 2);
  float*    sc   = (float*)alloc((size_t)Bn * Hn * Nn * Nn * 4);
  _Float16* alph = (_Float16*)alloc((size_t)Bn * Hn * Nn * Nn * 2);
  float*    yF   = (float*)alloc((size_t)Bn * Nn * Dn * 4);
  _Float16* yH   = (_Float16*)alloc((size_t)Bn * Nn * Dn * 2);
  _Float16* t1   = (_Float16*)alloc((size_t)Bn * Nn * FFNn * 2);
  float*    y2   = (float*)alloc((size_t)Bn * Nn * Dn * 4);
  (void)ws_size; (void)n_in; (void)in_sizes; (void)out_size;

  k_prep_hidden<<<(Bn * Nn * Dn + 255) / 256, 256, 0, stream>>>(
      hidden, a0, a1, a2, a3, hh, ht, hs);
  k_prep_w<<<(5 * 128 * 128 + 255) / 256, 256, 0, stream>>>(
      wq, wk, wv, w1, w2, wqT, wkT, wvT, w1T, w2T);
  k_gat<<<dim3(Nn / 16, Bn), 256, 0, stream>>>(hh, hs, adj, attL);
  k_softmax<<<(Bn * Nn) / 8, 256, 0, stream>>>(attL, attH);
  k_local<<<dim3(Nn / 16, Bn), 256, 0, stream>>>(attH, ht, locF, locH);
  k_aent<<<Bn, 32, 0, stream>>>(locF, wa, ba, aent);
  k_qkv<<<dim3(Nn / 16, Bn), 256, 0, stream>>>(locH, wqT, wkT, wvT, bq, bk, bv,
                                               qh, kh, vt);
  k_scores<<<dim3(Nn / 16, Hn, Bn), 256, 0, stream>>>(qh, kh, seq_mask, sc);
  k_entmax<<<(Bn * Hn * Nn) / 8, 256, 0, stream>>>(sc, aent, alph);
  k_y<<<dim3(Nn / 16, Bn), 256, 0, stream>>>(alph, vt, yF, yH);
  k_ffn1<<<dim3(Nn / 16, Bn), 256, 0, stream>>>(yH, w1T, b1, t1);
  k_ffn2<<<dim3(Nn / 16, Bn), 256, 0, stream>>>(t1, w2T, b2, y2);
  k_final<<<(Bn * Nn) / 8, 256, 0, stream>>>(locF, yF, y2, ln_g, ln_b, out);
}
